// MultiHeadAttention_5188320494258
// MI455X (gfx1250) — compile-verified, matrix operand order checked by reading, not run on hardware
//
#include <hip/hip_runtime.h>
#include <hip/hip_bf16.h>

// ---------------------------------------------------------------------------
// MI455X (gfx1250) fused multi-head attention with learned temporal/distance
// bias.  bf16 WMMA (v_wmma_f32_16x16x32_bf16) for all GEMMs, fp32 accumulate,
// flash-style online softmax, fused bias/mask plane, TDM (tensor_load_to_lds)
// staging for the input-projection B tiles and async global->LDS
// (global_load_async_to_lds_b128 + s_wait_asynccnt) for the output projection.
// ---------------------------------------------------------------------------

#define BATCH 4
#define SEQ   1024
#define DMODEL 512
#define HEADS 8
#define DK    64

typedef __attribute__((ext_vector_type(16))) __bf16       v16bf;
typedef __attribute__((ext_vector_type(2)))  __bf16       v2bf;
typedef __attribute__((ext_vector_type(8)))  float        v8f;
typedef __attribute__((ext_vector_type(8)))  unsigned int v8u;
typedef __attribute__((ext_vector_type(4)))  unsigned int v4u;
typedef __attribute__((ext_vector_type(8)))  int          v8i;
typedef __attribute__((ext_vector_type(4)))  int          v4i;

#if __has_builtin(__builtin_amdgcn_tensor_load_to_lds)
#define HAVE_TDM 1
#endif

// Native converts: compiler lowers to v_cvt_pk_bf16_f32 / scalar cvt.
__device__ __forceinline__ unsigned short f2bf(float x) {
  return __builtin_bit_cast(unsigned short, (__bf16)x);
}
__device__ __forceinline__ unsigned int pack_bf16(float x, float y) {
  v2bf t; t[0] = (__bf16)x; t[1] = (__bf16)y;
  return __builtin_bit_cast(unsigned int, t);
}

// A/B fragment loader for v_wmma_*_16x16x32_bf16.
// Tile is [16 rows][32 K] row-major (ld elements between rows), bf16.
// Lane l<16: row=l, K 0..7 and 16..23 ; lane l>=16: row=l-16, K 8..15, 24..31.
__device__ __forceinline__ v16bf load_frag_bf16(const unsigned short* p, int ld) {
  int lane = threadIdx.x & 31;
  const unsigned short* q = p + (size_t)(lane & 15) * ld + ((lane >> 4) << 3);
  uint4 a = *reinterpret_cast<const uint4*>(q);
  uint4 b = *reinterpret_cast<const uint4*>(q + 16);
  v8u u;
  u[0]=a.x; u[1]=a.y; u[2]=a.z; u[3]=a.w;
  u[4]=b.x; u[5]=b.y; u[6]=b.z; u[7]=b.w;
  return __builtin_bit_cast(v16bf, u);
}

// Same fragment, but source is fp32 in global memory (converted inline).
__device__ __forceinline__ v16bf load_frag_f32(const float* p, int ld) {
  int lane = threadIdx.x & 31;
  const float* q = p + (size_t)(lane & 15) * ld + ((lane >> 4) << 3);
  float4 f0 = *reinterpret_cast<const float4*>(q);
  float4 f1 = *reinterpret_cast<const float4*>(q + 4);
  float4 f2 = *reinterpret_cast<const float4*>(q + 16);
  float4 f3 = *reinterpret_cast<const float4*>(q + 20);
  v8u u;
  u[0]=pack_bf16(f0.x,f0.y); u[1]=pack_bf16(f0.z,f0.w);
  u[2]=pack_bf16(f1.x,f1.y); u[3]=pack_bf16(f1.z,f1.w);
  u[4]=pack_bf16(f2.x,f2.y); u[5]=pack_bf16(f2.z,f2.w);
  u[6]=pack_bf16(f3.x,f3.y); u[7]=pack_bf16(f3.z,f3.w);
  return __builtin_bit_cast(v16bf, u);
}

__device__ __forceinline__ v8f wmma_bf16(v16bf a, v16bf b, v8f c) {
  return __builtin_amdgcn_wmma_f32_16x16x32_bf16(false, a, false, b, (short)0, c,
                                                 false, false);
}

// CDNA5 async copy: 16 bytes global -> LDS, tracked by ASYNCcnt.
__device__ __forceinline__ void async_g2l_b128(unsigned ldsOff, const void* gptr) {
  unsigned long long ga = (unsigned long long)(uintptr_t)gptr;
  asm volatile("global_load_async_to_lds_b128 %0, %1, off"
               :: "v"(ldsOff), "v"(ga) : "memory");
}
__device__ __forceinline__ void wait_async0() {
  asm volatile("s_wait_asynccnt 0x0" ::: "memory");
}

// CDNA5 Tensor Data Mover: one 2D bf16 tile (tileK x tileN elements, row
// stride `stride` elements) global -> LDS, tracked by TENSORcnt.
// D# layout per CDNA5 ISA.  This toolchain's
// builtin takes 6 args: (v4u g0, v8i g1, v4i g2, v4i g3, v8i g4, i32 cpol).
__device__ __forceinline__ void tdm_load_tile_bf16(unsigned ldsAddr,
                                                   const void* gptr,
                                                   unsigned tileK, unsigned tileN,
                                                   unsigned dim0, unsigned dim1,
                                                   unsigned stride) {
#if defined(HAVE_TDM)
  unsigned long long ga = (unsigned long long)(uintptr_t)gptr;
  v4u g0;
  g0[0] = 1u;                                            // count=1, user mode
  g0[1] = ldsAddr;                                       // lds_addr (bytes)
  g0[2] = (unsigned)(ga & 0xFFFFFFFFu);                  // global_addr[31:0]
  g0[3] = (unsigned)((ga >> 32) & 0x01FFFFFFu) | (2u << 30);  // addr[56:32]|type=2
  v8i g1;
  g1[0] = (int)(1u << 16);                               // data_size=1 (2 bytes)
  g1[1] = (int)((dim0 & 0xFFFFu) << 16);                 // tensor_dim0 lo16
  g1[2] = (int)((dim0 >> 16) | ((dim1 & 0xFFFFu) << 16));// dim0 hi / dim1 lo
  g1[3] = (int)((dim1 >> 16) | (tileK << 16));           // dim1 hi / tile_dim0
  g1[4] = (int)(tileN);                                  // tile_dim1, tile_dim2=0
  g1[5] = (int)(stride);                                 // tensor_dim0_stride lo
  g1[6] = 0;                                             // stride hi / dim1_stride
  g1[7] = 0;
  v4i z4 = {};
  v8i z8 = {};
  __builtin_amdgcn_tensor_load_to_lds(g0, g1, z4, z4, z8, 0);
  __builtin_amdgcn_s_wait_tensorcnt(0);
#endif
}

// ---------------------------------------------------------------------------
// One-time weight conversion: W fp32 [K][N] -> Wt bf16 [N][K] (transposed so
// GEMM B-fragments are two contiguous 16B loads).  blockIdx.y selects matrix.
// ---------------------------------------------------------------------------
__global__ __launch_bounds__(256) void wconv_kernel(const float* __restrict__ W0,
                                                    const float* __restrict__ W1,
                                                    const float* __restrict__ W2,
                                                    const float* __restrict__ W3,
                                                    unsigned short* __restrict__ out) {
  const float* srcs[4] = {W0, W1, W2, W3};
  const float* src = srcs[blockIdx.y];
  unsigned short* dst = out + (size_t)blockIdx.y * DMODEL * DMODEL;
  int e = blockIdx.x * 256 + threadIdx.x;     // 0 .. 512*512-1
  int k = e >> 9, n = e & 511;
  dst[(size_t)n * DMODEL + k] = f2bf(src[(size_t)k * DMODEL + n]);
}

// ---------------------------------------------------------------------------
// Input projection: Y = X @ W + bias, X [4096x512] fp32, Wt bf16 [N][K].
// Output bf16, scaled, with head-split layout:
//   mode 0: out[((b*H+h)*S + s)*dk + d]       (q, k)
//   mode 1: out[((b*H+h)*dk + d)*S + s]       (v, transposed for PV B-frags)
// Block = 256 threads = 8 waves (2 M x 4 N), block tile 64x128, wave tile 32x32.
// B tile staged to LDS by the Tensor Data Mover (one descriptor per K-step).
// ---------------------------------------------------------------------------
__global__ __launch_bounds__(256) void proj_kernel(const float* __restrict__ X,
                                                   const unsigned short* __restrict__ Wt,
                                                   const float* __restrict__ bias,
                                                   unsigned short* __restrict__ out,
                                                   int mode, float scale) {
  __shared__ __align__(16) unsigned short Bt[128 * 32];  // [n][k] bf16
  const int tid = threadIdx.x;
  const int wv = tid >> 5, lane = tid & 31;
  const int waveM = wv & 1, waveN = wv >> 1;
  const int rowBase = blockIdx.x * 64;
  const int colBase = blockIdx.y * 128;

  v8f acc[2][2] = {};
  for (int k0 = 0; k0 < DMODEL; k0 += 32) {
#if defined(HAVE_TDM)
    if (wv == 0) {  // wave 0 drives the TDM for the whole 128x32 bf16 tile
      tdm_load_tile_bf16((unsigned)(uintptr_t)(&Bt[0]),
                         Wt + (size_t)colBase * DMODEL + k0,
                         /*tileK=*/32, /*tileN=*/128,
                         /*dim0=*/DMODEL, /*dim1=*/DMODEL, /*stride=*/DMODEL);
    }
#else
    {  // fallback: async-stage 128x32 bf16 B tile, 2 x 16B per thread
      int row = tid >> 1, half = tid & 1;
      const unsigned short* gp =
          Wt + (size_t)(colBase + row) * DMODEL + k0 + half * 16;
      unsigned l0 = (unsigned)(uintptr_t)(&Bt[row * 32 + half * 16]);
      async_g2l_b128(l0, gp);
      async_g2l_b128(l0 + 16u, gp + 8);
    }
    wait_async0();
#endif
    __syncthreads();
    v16bf a0 = load_frag_f32(X + (size_t)(rowBase + waveM * 32)      * DMODEL + k0, DMODEL);
    v16bf a1 = load_frag_f32(X + (size_t)(rowBase + waveM * 32 + 16) * DMODEL + k0, DMODEL);
    v16bf b0 = load_frag_bf16(Bt + (waveN * 32)      * 32, 32);
    v16bf b1 = load_frag_bf16(Bt + (waveN * 32 + 16) * 32, 32);
    acc[0][0] = wmma_bf16(a0, b0, acc[0][0]);
    acc[0][1] = wmma_bf16(a0, b1, acc[0][1]);
    acc[1][0] = wmma_bf16(a1, b0, acc[1][0]);
    acc[1][1] = wmma_bf16(a1, b1, acc[1][1]);
    __syncthreads();
  }

  const int rofs = (lane >> 4) << 3;
  const int cl = lane & 15;
#pragma unroll
  for (int i = 0; i < 2; ++i)
#pragma unroll
    for (int j = 0; j < 2; ++j)
#pragma unroll
      for (int r = 0; r < 8; ++r) {
        int row = rowBase + waveM * 32 + i * 16 + r + rofs;
        int col = colBase + waveN * 32 + j * 16 + cl;
        float vv = (acc[i][j][r] + bias[col]) * scale;
        int bb = row >> 10, s = row & 1023;
        int h = col >> 6, d = col & 63;
        size_t idx = (mode == 0)
            ? ((((size_t)bb * HEADS + h) * SEQ + s) * DK + d)
            : ((((size_t)bb * HEADS + h) * DK + d) * SEQ + s);
        out[idx] = f2bf(vv);
      }
}

// ---------------------------------------------------------------------------
// Fused bias plane: biasM[b,i,j] = mask ? -1e9 : td_w . relu-mix(t,d) + td_b.
// Avoids materializing the 1 GB [B,S,S,64] intermediates.
// ---------------------------------------------------------------------------
__global__ __launch_bounds__(256) void bias_kernel(const float* __restrict__ tm,
                                                   const float* __restrict__ dm,
                                                   const int*   __restrict__ mask,
                                                   const float* __restrict__ tm_w,
                                                   const float* __restrict__ tm_b,
                                                   const float* __restrict__ dm_w,
                                                   const float* __restrict__ dm_b,
                                                   const float* __restrict__ td_w,
                                                   const float* __restrict__ td_b,
                                                   float* __restrict__ outB, int n) {
  int i = blockIdx.x * 256 + threadIdx.x;
  if (i >= n) return;
  if (mask[i] == 1) { outB[i] = -1e9f; return; }
  const float E = 2.718281828459045f;
  float t = 1.0f / __logf(E + tm[i]);
  float d = 1.0f / __logf(E + dm[i]);
  float acc = 0.f;
#pragma unroll 8
  for (int c = 0; c < 64; ++c) {
    float th = fmaxf(fmaf(t, tm_w[c], tm_b[c]), 0.f);
    float dh = fmaxf(fmaf(d, dm_w[c], dm_b[c]), 0.f);
    acc = fmaf(td_w[c], 0.5f * dh + 0.5f * th, acc);
  }
  outB[i] = acc + td_b[0];
}

// ---------------------------------------------------------------------------
// Flash attention: one wave owns 16 queries x full dk=64 output for one (b,h).
// Key loop in blocks of 32: 4 score WMMAs (K=dk, split 2x32), bias add,
// online softmax with wave32 shfl_xor row reductions, probs -> LDS -> A frag,
// 4 PV WMMAs against transposed V.  Scores scale folded into q projection.
// ---------------------------------------------------------------------------
__global__ __launch_bounds__(128) void attn_kernel(const unsigned short* __restrict__ qb,
                                                   const unsigned short* __restrict__ kb,
                                                   const unsigned short* __restrict__ vb,
                                                   const float* __restrict__ biasM,
                                                   unsigned short* __restrict__ attn) {
  __shared__ __align__(16) unsigned short probs[4][16 * 32];
  const int lane = threadIdx.x & 31, w = threadIdx.x >> 5;
  const int wid = blockIdx.x * 4 + w;
  const int b = wid >> 9;            // 512 waves per batch
  const int h = (wid >> 6) & 7;
  const int q0 = (wid & 63) << 4;

  const size_t bh = (size_t)b * HEADS + h;
  const unsigned short* qp = qb + (bh * SEQ + q0) * DK;
  const unsigned short* kp = kb + bh * SEQ * DK;
  const unsigned short* vp = vb + bh * DK * SEQ;
  const float* bp = biasM + ((size_t)b * SEQ + q0) * SEQ;

  v16bf aq0 = load_frag_bf16(qp + 0,  DK);   // dk 0..31
  v16bf aq1 = load_frag_bf16(qp + 32, DK);   // dk 32..63

  v8f acc[4] = {};
  float m[8], l[8];
#pragma unroll
  for (int r = 0; r < 8; ++r) { m[r] = -1e30f; l[r] = 0.f; }
  const int rofs = (lane >> 4) << 3;
  const int cl = lane & 15;

  for (int k0 = 0; k0 < SEQ; k0 += 32) {
    if (k0 + 32 < SEQ) {  // pull next K/V tiles toward the WGP
      __builtin_prefetch(kp + (size_t)(k0 + 32) * DK, 0, 1);
      __builtin_prefetch(vp + (size_t)(k0 + 32), 0, 1);
    }
    // ---- scores: 16 q x 32 keys, K-dim = dk = 64 (two 32-chunks) ----
    v8f s0 = {}, s1 = {};
    v16bf bk0 = load_frag_bf16(kp + (size_t)k0 * DK,             DK);
    v16bf bk1 = load_frag_bf16(kp + (size_t)(k0 + 16) * DK,      DK);
    v16bf bk2 = load_frag_bf16(kp + (size_t)k0 * DK + 32,        DK);
    v16bf bk3 = load_frag_bf16(kp + (size_t)(k0 + 16) * DK + 32, DK);
    s0 = wmma_bf16(aq0, bk0, s0);
    s1 = wmma_bf16(aq0, bk1, s1);
    s0 = wmma_bf16(aq1, bk2, s0);
    s1 = wmma_bf16(aq1, bk3, s1);

    // ---- add precomputed (bias + mask) plane ----
#pragma unroll
    for (int r = 0; r < 8; ++r) {
      const float* bpr = bp + (size_t)(r + rofs) * SEQ + k0 + cl;
      s0[r] += bpr[0];
      s1[r] += bpr[16];
    }

    // ---- online softmax; row stats reduced over 16 lanes (wave32) ----
#pragma unroll
    for (int r = 0; r < 8; ++r) {
      float mx = fmaxf(s0[r], s1[r]);
      mx = fmaxf(mx, __shfl_xor(mx, 1, 32));
      mx = fmaxf(mx, __shfl_xor(mx, 2, 32));
      mx = fmaxf(mx, __shfl_xor(mx, 4, 32));
      mx = fmaxf(mx, __shfl_xor(mx, 8, 32));
      float mnew  = fmaxf(m[r], mx);
      float alpha = __expf(m[r] - mnew);
      float p0 = __expf(s0[r] - mnew);
      float p1 = __expf(s1[r] - mnew);
      float rs = p0 + p1;
      rs += __shfl_xor(rs, 1, 32);
      rs += __shfl_xor(rs, 2, 32);
      rs += __shfl_xor(rs, 4, 32);
      rs += __shfl_xor(rs, 8, 32);
      l[r] = l[r] * alpha + rs;
      m[r] = mnew;
#pragma unroll
      for (int t = 0; t < 4; ++t) acc[t][r] *= alpha;
      int row = r + rofs;
      probs[w][row * 32 + cl]      = f2bf(p0);
      probs[w][row * 32 + cl + 16] = f2bf(p1);
    }

    // ---- PV: probs (16x32) @ v (32x64), v is [d][s] so B-frags are direct ----
    v16bf ap = load_frag_bf16(&probs[w][0], 32);
#pragma unroll
    for (int t = 0; t < 4; ++t) {
      v16bf bvf = load_frag_bf16(vp + (size_t)(t * 16) * SEQ + k0, SEQ);
      acc[t] = wmma_bf16(ap, bvf, acc[t]);
    }
  }

  // ---- normalize and store bf16 [B,S,D] for the output projection ----
#pragma unroll
  for (int r = 0; r < 8; ++r) {
    float inv = 1.0f / l[r];
    int row = q0 + r + rofs;
    unsigned short* op = attn + ((size_t)b * SEQ + row) * DMODEL + h * DK;
#pragma unroll
    for (int t = 0; t < 4; ++t) op[t * 16 + cl] = f2bf(acc[t][r] * inv);
  }
}

// ---------------------------------------------------------------------------
// Output projection: attn(bf16) [4096x512] @ Wo + bo -> fp32 d_out.
// A fragments load straight from global bf16; Wo (bf16 [N][K]) async-staged
// with global_load_async_to_lds_b128 (ASYNCcnt path).
// ---------------------------------------------------------------------------
__global__ __launch_bounds__(256) void outproj_kernel(const unsigned short* __restrict__ X,
                                                      const unsigned short* __restrict__ Wt,
                                                      const float* __restrict__ bias,
                                                      float* __restrict__ out) {
  __shared__ __align__(16) unsigned short Bt[128 * 32];
  const int tid = threadIdx.x;
  const int wv = tid >> 5, lane = tid & 31;
  const int waveM = wv & 1, waveN = wv >> 1;
  const int rowBase = blockIdx.x * 64;
  const int colBase = blockIdx.y * 128;

  v8f acc[2][2] = {};
  for (int k0 = 0; k0 < DMODEL; k0 += 32) {
    {
      int row = tid >> 1, half = tid & 1;
      const unsigned short* gp =
          Wt + (size_t)(colBase + row) * DMODEL + k0 + half * 16;
      unsigned l0 = (unsigned)(uintptr_t)(&Bt[row * 32 + half * 16]);
      async_g2l_b128(l0, gp);
      async_g2l_b128(l0 + 16u, gp + 8);
    }
    wait_async0();
    __syncthreads();
    v16bf a0 = load_frag_bf16(X + (size_t)(rowBase + waveM * 32)      * DMODEL + k0, DMODEL);
    v16bf a1 = load_frag_bf16(X + (size_t)(rowBase + waveM * 32 + 16) * DMODEL + k0, DMODEL);
    v16bf b0 = load_frag_bf16(Bt + (waveN * 32)      * 32, 32);
    v16bf b1 = load_frag_bf16(Bt + (waveN * 32 + 16) * 32, 32);
    acc[0][0] = wmma_bf16(a0, b0, acc[0][0]);
    acc[0][1] = wmma_bf16(a0, b1, acc[0][1]);
    acc[1][0] = wmma_bf16(a1, b0, acc[1][0]);
    acc[1][1] = wmma_bf16(a1, b1, acc[1][1]);
    __syncthreads();
  }

  const int rofs = (lane >> 4) << 3;
  const int cl = lane & 15;
#pragma unroll
  for (int i = 0; i < 2; ++i)
#pragma unroll
    for (int j = 0; j < 2; ++j)
#pragma unroll
      for (int r = 0; r < 8; ++r) {
        int row = rowBase + waveM * 32 + i * 16 + r + rofs;
        int col = colBase + waveN * 32 + j * 16 + cl;
        out[(size_t)row * DMODEL + col] = acc[i][j][r] + bias[col];
      }
}

// ---------------------------------------------------------------------------
extern "C" void kernel_launch(void* const* d_in, const int* in_sizes, int n_in,
                              void* d_out, int out_size, void* d_ws, size_t ws_size,
                              hipStream_t stream) {
  const float* Q    = (const float*)d_in[0];
  const float* K    = (const float*)d_in[1];
  const float* V    = (const float*)d_in[2];
  const float* tm   = (const float*)d_in[3];
  const float* dm   = (const float*)d_in[4];
  const int*   mask = (const int*)  d_in[5];
  const float* Wq   = (const float*)d_in[6];
  const float* bq   = (const float*)d_in[7];
  const float* Wk   = (const float*)d_in[8];
  const float* bk   = (const float*)d_in[9];
  const float* Wv   = (const float*)d_in[10];
  const float* bv   = (const float*)d_in[11];
  const float* Wo   = (const float*)d_in[12];
  const float* bo   = (const float*)d_in[13];
  const float* tm_w = (const float*)d_in[14];
  const float* tm_b = (const float*)d_in[15];
  const float* dm_w = (const float*)d_in[16];
  const float* dm_b = (const float*)d_in[17];
  const float* td_w = (const float*)d_in[18];
  const float* td_b = (const float*)d_in[19];

  char* ws = (char*)d_ws;
  unsigned short* qbuf = (unsigned short*)(ws);                 // 4 MB bf16 [B,H,S,dk]
  unsigned short* kbuf = (unsigned short*)(ws + (4u  << 20));   // 4 MB bf16 [B,H,S,dk]
  unsigned short* vbuf = (unsigned short*)(ws + (8u  << 20));   // 4 MB bf16 [B,H,dk,S]
  unsigned short* abuf = (unsigned short*)(ws + (12u << 20));   // 4 MB bf16 [B,S,D]
  float*          bias = (float*)         (ws + (16u << 20));   // 16 MB fp32 [B,S,S]
  unsigned short* wt   = (unsigned short*)(ws + (32u << 20));   // 2 MB bf16 4x[N][K]
  const size_t WSZ = (size_t)DMODEL * DMODEL;

  wconv_kernel<<<dim3(1024, 4), 256, 0, stream>>>(Wq, Wk, Wv, Wo, wt);

  dim3 gproj(64, 4);
  proj_kernel<<<gproj, 256, 0, stream>>>(Q, wt + 0 * WSZ, bq, qbuf, 0, 0.125f);
  proj_kernel<<<gproj, 256, 0, stream>>>(K, wt + 1 * WSZ, bk, kbuf, 0, 1.0f);
  proj_kernel<<<gproj, 256, 0, stream>>>(V, wt + 2 * WSZ, bv, vbuf, 1, 1.0f);
  bias_kernel<<<16384, 256, 0, stream>>>(tm, dm, mask, tm_w, tm_b, dm_w, dm_b,
                                         td_w, td_b, bias, BATCH * SEQ * SEQ);
  attn_kernel<<<512, 128, 0, stream>>>(qbuf, kbuf, vbuf, bias, abuf);
  outproj_kernel<<<gproj, 256, 0, stream>>>(abuf, wt + 3 * WSZ, bo, (float*)d_out);
}
